// EnhancedBalSCL_52381421142523
// MI455X (gfx1250) — matrix operand order repaired and checked by reading, not compiled, not run on hardware
//
#include <hip/hip_runtime.h>
#include <hip/hip_bf16.h>

// ---------------------------------------------------------------------------
// EnhancedBalSCL loss for MI455X (gfx1250, wave32, WMMA).
// Fused: GEMM (bf16 hi/lo split, 3x v_wmma_f32_16x16x32_bf16) + exp/log
// epilogue + deterministic partial-sum reductions. No float atomics.
// Double-buffered LDS with gfx1250 async global->LDS copies
// (GLOBAL_LOAD_ASYNC_TO_LDS_B128 / ASYNCcnt) to overlap copy with WMMA.
// ---------------------------------------------------------------------------

typedef __attribute__((ext_vector_type(16))) __bf16 v16bf;
typedef __attribute__((ext_vector_type(8)))  float  v8f;

#define NB   4096      // batch
#define ND   1024      // feature dim
#define NC   1000      // classes
#define NCP  1024      // classes padded to tile multiple
#define NKT  (ND / 32) // K tiles
#define INV_TAU 10.0f
#define EPSV 1e-8f

#define GEMM1_SLOTS 64               // 32 col-tiles x 2 n-strips
#define GEMM2_SLOTS 16               // 8 col-tiles x 2 n-strips
#define DEN_SLOTS   (GEMM1_SLOTS + GEMM2_SLOTS)   // 80

// -------- gfx1250 async copy path (probe via __has_builtin) --------
#if defined(__HIP_DEVICE_COMPILE__) && defined(__gfx1250__) && \
    __has_builtin(__builtin_amdgcn_global_load_async_to_lds_b128)
#define ASYNC_COPY 1
#else
#define ASYNC_COPY 0
#endif

static __device__ inline void async_b128(const void* g, void* l) {
#if ASYNC_COPY
    // builtin signature (from probe): (v4i AS1*, v4i AS3*, Imm int, Imm cpol)
    typedef int v4i_ __attribute__((vector_size(16)));
    typedef __attribute__((address_space(1))) v4i_* gp_t;
    typedef __attribute__((address_space(3))) v4i_* lp_t;
    // generic LDS pointer low 32 bits == LDS byte offset (flat aperture rule)
    __builtin_amdgcn_global_load_async_to_lds_b128(
        (gp_t)(uintptr_t)g, (lp_t)(uint32_t)(uintptr_t)l, 0, 0);
#else
    *(uint4*)l = *(const uint4*)g;
#endif
}

static __device__ inline void wait_async0() {
#if ASYNC_COPY
#if __has_builtin(__builtin_amdgcn_s_wait_asynccnt)
    __builtin_amdgcn_s_wait_asynccnt(0);
#else
    asm volatile("s_wait_asynccnt 0x0" ::: "memory");
#endif
#endif
}

struct Frag32B { uint4 q0, q1; };

static __device__ inline v16bf make_frag(uint4 a, uint4 b) {
    Frag32B f{a, b};
    return __builtin_bit_cast(v16bf, f);
}

static __device__ inline v8f wmma_bf16(v16bf a, v16bf b, v8f c) {
    return __builtin_amdgcn_wmma_f32_16x16x32_bf16(false, a, false, b,
                                                   (short)0, c, false, false);
}

// ---------------- setup kernels ----------------

__global__ __launch_bounds__(256) void k_zero(int* counts, float* ctr) {
    int i = blockIdx.x * 256 + threadIdx.x;
    if (i < NCP) counts[i] = 0;
    if (i < NB)  ctr[i] = 0.0f;
}

__global__ __launch_bounds__(256) void k_hist(const int* __restrict__ t, int* counts) {
    int i = blockIdx.x * 256 + threadIdx.x;
    if (i < NB) atomicAdd(&counts[t[i]], 1);
}

__global__ __launch_bounds__(256) void k_weights(const int* __restrict__ t,
                                                 const int* __restrict__ counts,
                                                 float* __restrict__ w,
                                                 float* __restrict__ winv) {
    int i = blockIdx.x * 256 + threadIdx.x;
    if (i < NB)  w[i] = 1.0f / (float)(counts[t[i]] + 1);
    if (i < NCP) winv[i] = (i < NC) ? 1.0f / (float)(counts[i] + 1) : 0.0f;
}

// f32 -> (bf16 hi, bf16 lo) split
__global__ __launch_bounds__(256) void k_cvt_feat(const float* __restrict__ src,
                                                  __bf16* __restrict__ hi,
                                                  __bf16* __restrict__ lo, int n) {
    int i = blockIdx.x * 256 + threadIdx.x;
    if (i < n) {
        float x = src[i];
        __bf16 h = (__bf16)x;
        hi[i] = h;
        lo[i] = (__bf16)(x - (float)h);
    }
}

// centers: rows [0,1000) converted, rows [1000,1024) zero-padded
__global__ __launch_bounds__(256) void k_cvt_centers(const float* __restrict__ src,
                                                     __bf16* __restrict__ hi,
                                                     __bf16* __restrict__ lo) {
    int i = blockIdx.x * 256 + threadIdx.x;   // over NCP*ND
    int row = i >> 10;
    float x = (row < NC) ? src[i] : 0.0f;     // src is [1000][1024] contiguous
    __bf16 h = (__bf16)x;
    hi[i] = h;
    lo[i] = (__bf16)(x - (float)h);
}

// ---------------- fused WMMA GEMM + epilogue ----------------
// 128x128 output tile per workgroup; 8 waves in a 4(M) x 2(N) grid,
// each wave owns a 32x64 region = 2x4 subtiles of 16x16.
// A rows: features[rowBase..]; B rows: features/centers[colBase..] (B = row^T).

template <bool CENTER_PASS>
__global__ __launch_bounds__(256)
void k_gemm(const __bf16* __restrict__ Ahi, const __bf16* __restrict__ Alo,
            const __bf16* __restrict__ Bhi, const __bf16* __restrict__ Blo,
            const int*   __restrict__ targets,
            const float* __restrict__ colw,      // w (GEMM1) or winv (GEMM2)
            float* __restrict__ denomPart,       // [NB][DEN_SLOTS]
            float* __restrict__ posPart,         // [NB][GEMM1_SLOTS]
            float* __restrict__ ctr)             // [NB]
{
    // [buf][hi=0/lo=1][row][k(+8 pad -> 80B row stride)]
    __shared__ __bf16 sA[2][2][128][40];
    __shared__ __bf16 sB[2][2][128][40];

    const int tid    = threadIdx.x;
    const int wave   = tid >> 5;
    const int lane   = tid & 31;
    const int h      = lane >> 4;     // half-wave
    const int m      = lane & 15;
    const int mStrip = wave & 3;      // row strip (x32)
    const int nStrip = wave >> 2;     // col strip (x64)
    const int rowBase = blockIdx.y * 128;
    const int colBase = blockIdx.x * 128;

    v8f acc[2][4] = {};

    // stage one 32-wide K slice (A & B, hi & lo) into LDS buffer `buf`
    auto stage = [&](int kt, int buf) {
#pragma unroll
        for (int t = 0; t < 2; ++t) {
            int idx = tid + t * 256;          // 0..511
            int r   = idx >> 2;               // 0..127
            int kc  = (idx & 3) * 8;          // 0,8,16,24
            size_t ga = (size_t)(rowBase + r) * ND + kt + kc;
            size_t gb = (size_t)(colBase + r) * ND + kt + kc;
            async_b128(&Ahi[ga], &sA[buf][0][r][kc]);
            async_b128(&Alo[ga], &sA[buf][1][r][kc]);
            async_b128(&Bhi[gb], &sB[buf][0][r][kc]);
            async_b128(&Blo[gb], &sB[buf][1][r][kc]);
        }
    };

    auto compute = [&](int buf) {
        // A fragments: lane (h,m) holds row m, K = {8h..8h+7, 16+8h..16+8h+7}
        v16bf aHi[2], aLo[2];
#pragma unroll
        for (int sm = 0; sm < 2; ++sm) {
            int ar = mStrip * 32 + sm * 16 + m;
            aHi[sm] = make_frag(*(const uint4*)&sA[buf][0][ar][8 * h],
                                *(const uint4*)&sA[buf][0][ar][16 + 8 * h]);
            aLo[sm] = make_frag(*(const uint4*)&sA[buf][1][ar][8 * h],
                                *(const uint4*)&sA[buf][1][ar][16 + 8 * h]);
        }
        // B fragments: lane (h,m) holds col m, K = {8h..8h+7, 16+8h..16+8h+7}
#pragma unroll
        for (int sn = 0; sn < 4; ++sn) {
            int br = nStrip * 64 + sn * 16 + m;
            v16bf bHi = make_frag(*(const uint4*)&sB[buf][0][br][8 * h],
                                  *(const uint4*)&sB[buf][0][br][16 + 8 * h]);
            v16bf bLo = make_frag(*(const uint4*)&sB[buf][1][br][8 * h],
                                  *(const uint4*)&sB[buf][1][br][16 + 8 * h]);
#pragma unroll
            for (int sm = 0; sm < 2; ++sm) {
                acc[sm][sn] = wmma_bf16(aHi[sm], bHi, acc[sm][sn]);
                acc[sm][sn] = wmma_bf16(aHi[sm], bLo, acc[sm][sn]);
                acc[sm][sn] = wmma_bf16(aLo[sm], bHi, acc[sm][sn]);
            }
        }
    };

    // software pipeline: copy of tile k+1 overlaps WMMA of tile k
    stage(0, 0);
#pragma unroll 2
    for (int kt = 0; kt < NKT; ++kt) {
        int cur = kt & 1;
        wait_async0();        // own async copies into buf[cur] complete
        __syncthreads();      // all waves' copies done; buf[cur^1] free to reuse
        if (kt + 1 < NKT) stage((kt + 1) * 32, cur ^ 1);
        compute(cur);
    }

    // ---- epilogue: exp / weighted denom / positive-log mask / center gather ----
    int   cols[4];
    float wc[4];
    int   tc[4];
#pragma unroll
    for (int sn = 0; sn < 4; ++sn) {
        int col  = colBase + nStrip * 64 + sn * 16 + m;
        cols[sn] = col;
        wc[sn]   = colw[col];
        tc[sn]   = CENTER_PASS ? 0 : targets[col];
    }
    const int slot = CENTER_PASS ? (GEMM1_SLOTS + blockIdx.x * 2 + nStrip)
                                 : (blockIdx.x * 2 + nStrip);

#pragma unroll
    for (int sm = 0; sm < 2; ++sm) {
#pragma unroll
        for (int r = 0; r < 8; ++r) {
            int row  = rowBase + mStrip * 32 + sm * 16 + 8 * h + r;
            int trow = targets[row];
            float dAcc = 0.0f, pAcc = 0.0f;
#pragma unroll
            for (int sn = 0; sn < 4; ++sn) {
                float s = acc[sm][sn][r] * INV_TAU;
                float e = __expf(s);
                dAcc += e * wc[sn];
                if (!CENTER_PASS) {
                    if (tc[sn] == trow && cols[sn] != row)
                        pAcc += __logf(e + EPSV);
                } else {
                    if (cols[sn] == trow)
                        ctr[row] = __logf(e + EPSV);   // unique writer per row
                }
            }
            // reduce across the 16 lanes of this half-wave (same h)
#pragma unroll
            for (int off = 1; off < 16; off <<= 1) {
                dAcc += __shfl_xor(dAcc, off, 32);
                if (!CENTER_PASS) pAcc += __shfl_xor(pAcc, off, 32);
            }
            if (m == 0) {
                denomPart[row * DEN_SLOTS + slot] = dAcc;
                if (!CENTER_PASS) posPart[row * GEMM1_SLOTS + slot] = pAcc;
            }
        }
    }
}

// ---------------- finalize: deterministic fixed-order reduction ----------------

__global__ __launch_bounds__(256)
void k_finalize(const float* __restrict__ denomPart, const float* __restrict__ posPart,
                const float* __restrict__ ctr, const int* __restrict__ targets,
                const int* __restrict__ counts, float* __restrict__ out) {
    __shared__ float red[256];
    int tid = threadIdx.x;
    float local = 0.0f;
    for (int i = tid; i < NB; i += 256) {
        float denom = 0.0f;
        for (int s = 0; s < DEN_SLOTS; ++s) denom += denomPart[i * DEN_SLOTS + s];
        float pos = 0.0f;
        for (int s = 0; s < GEMM1_SLOTS; ++s) pos += posPart[i * GEMM1_SLOTS + s];
        float np = (float)counts[targets[i]];
        local += __logf(denom + EPSV) - (pos + ctr[i]) / np;
    }
    red[tid] = local;
    __syncthreads();
    for (int s = 128; s > 0; s >>= 1) {
        if (tid < s) red[tid] += red[tid + s];
        __syncthreads();
    }
    if (tid == 0) out[0] = red[0] / (float)NB;
}

// ---------------- host launcher ----------------

extern "C" void kernel_launch(void* const* d_in, const int* in_sizes, int n_in,
                              void* d_out, int out_size, void* d_ws, size_t ws_size,
                              hipStream_t stream) {
    (void)in_sizes; (void)n_in; (void)out_size; (void)ws_size;
    const float* d_centers  = (const float*)d_in[0];   // [1000,1024]
    const float* d_features = (const float*)d_in[1];   // [4096,1024]
    const int*   d_targets  = (const int*)d_in[2];     // [4096]

    char* p = (char*)d_ws;
    auto take = [&](size_t bytes) -> char* {
        char* r = p;
        p += (bytes + 255) & ~(size_t)255;
        return r;
    };
    __bf16* fhi = (__bf16*)take((size_t)NB * ND * 2);
    __bf16* flo = (__bf16*)take((size_t)NB * ND * 2);
    __bf16* chi = (__bf16*)take((size_t)NCP * ND * 2);
    __bf16* clo = (__bf16*)take((size_t)NCP * ND * 2);
    int*   counts    = (int*)take(NCP * 4);
    float* w         = (float*)take(NB * 4);
    float* winv      = (float*)take(NCP * 4);
    float* ctr       = (float*)take(NB * 4);
    float* denomPart = (float*)take((size_t)NB * DEN_SLOTS * 4);
    float* posPart   = (float*)take((size_t)NB * GEMM1_SLOTS * 4);

    k_zero   <<<16, 256, 0, stream>>>(counts, ctr);
    k_hist   <<<16, 256, 0, stream>>>(d_targets, counts);
    k_weights<<<16, 256, 0, stream>>>(d_targets, counts, w, winv);
    k_cvt_feat   <<<(NB * ND) / 256, 256, 0, stream>>>(d_features, fhi, flo, NB * ND);
    k_cvt_centers<<<(NCP * ND) / 256, 256, 0, stream>>>(d_centers, chi, clo);

    // GEMM1: features @ features^T  (4096x4096), 32x32 tiles of 128x128
    k_gemm<false><<<dim3(32, 32), 256, 0, stream>>>(fhi, flo, fhi, flo, d_targets,
                                                    w, denomPart, posPart, ctr);
    // GEMM2: features @ centers^T  (4096x1024 padded), 8x32 tiles
    k_gemm<true><<<dim3(8, 32), 256, 0, stream>>>(fhi, flo, chi, clo, d_targets,
                                                  winv, denomPart, posPart, ctr);

    k_finalize<<<1, 256, 0, stream>>>(denomPart, posPart, ctr, d_targets, counts,
                                      (float*)d_out);
}
